// matchingA_71983651881526
// MI455X (gfx1250) — matrix operand order repaired and checked by reading, not compiled, not run on hardware
//
#include <hip/hip_runtime.h>
#include <hip/hip_bf16.h>
#include <math.h>

// ---- problem constants (match reference) ----
constexpr int NN   = 1024;           // nodes
constexpr int SD   = 32;             // strainDim
constexpr int TD   = 128;            // timeDim
constexpr int CD   = 3;              // channels
constexpr int MIDD = 40;             // midLayer
constexpr int MC   = MIDD * CD;      // 120  (per-s projection width)
constexpr int KDIM = SD * MC;        // 3840 (collapsed GEMM reduction dim)
constexpr int GRP  = SD * CD;        // 96   (cosine groups)

typedef __attribute__((ext_vector_type(16))) _Float16 v16h;
typedef __attribute__((ext_vector_type(8)))  _Float16 v8h;
typedef __attribute__((ext_vector_type(8)))  float    v8f;

// ---------------------------------------------------------------------------
// Stage A: per node n, u = x_n @ Wu^T + bu, v = x_n @ Wv^T + bv,
// group norms over MID, fold Wnorm_w and 1/norm into f16 Uhat / Vhat.
// ---------------------------------------------------------------------------
__global__ void __launch_bounds__(128)
proj_norm_kernel(const float* __restrict__ x,
                 const float* __restrict__ Wu_w, const float* __restrict__ Wu_b,
                 const float* __restrict__ Wv_w, const float* __restrict__ Wv_b,
                 const float* __restrict__ Wnorm_w,
                 _Float16* __restrict__ Uh, _Float16* __restrict__ Vh)
{
    __shared__ float xs[SD * TD];     // 16 KB
    __shared__ float us[SD * MC];     // 15 KB
    __shared__ float vs[SD * MC];     // 15 KB
    __shared__ float su[GRP];
    __shared__ float sv[GRP];

    const int n   = blockIdx.x;
    const int tid = threadIdx.x;

    for (int i = tid; i < SD * TD; i += 128)
        xs[i] = x[(size_t)n * SD * TD + i];
    __syncthreads();

    // projection: 32x120 outputs per matrix, dot over T=128
    for (int o = tid; o < SD * MC; o += 128) {
        const int s = o / MC, m = o % MC;
        const float4* wu = (const float4*)(Wu_w + (size_t)m * TD);
        const float4* wv = (const float4*)(Wv_w + (size_t)m * TD);
        const float4* xr = (const float4*)(xs + s * TD);
        float au = 0.f, av = 0.f;
        #pragma unroll
        for (int t = 0; t < TD / 4; ++t) {
            const float4 xv = xr[t];
            const float4 a  = wu[t];
            const float4 b  = wv[t];
            au += xv.x * a.x + xv.y * a.y + xv.z * a.z + xv.w * a.w;
            av += xv.x * b.x + xv.y * b.y + xv.z * b.z + xv.w * b.w;
        }
        us[o] = au + Wu_b[m];
        vs[o] = av + Wv_b[m];
    }
    __syncthreads();

    // group norms: k = s*3 + c over us[s*120 + c*40 + 0..39]
    for (int g = tid; g < 2 * GRP; g += 128) {
        const int k = g % GRP;
        const int s = k / CD, c = k % CD;
        const float* p = ((g < GRP) ? us : vs) + s * MC + c * MIDD;
        float acc = 0.f;
        #pragma unroll
        for (int m2 = 0; m2 < MIDD; ++m2) acc += p[m2] * p[m2];
        const float inv = 1.0f / fmaxf(sqrtf(acc), 1e-4f);
        if (g < GRP) su[k] = Wnorm_w[k] * inv;   // fold Wnorm weight into U side
        else         sv[k] = inv;
    }
    __syncthreads();

    for (int o = tid; o < KDIM; o += 128) {
        const int s = o / MC, m = o % MC;
        const int k = s * CD + m / MIDD;
        Uh[(size_t)n * KDIM + o] = (_Float16)(us[o] * su[k]);
        Vh[(size_t)n * KDIM + o] = (_Float16)(vs[o] * sv[k]);
    }
}

// ---------------------------------------------------------------------------
// Stage B: Atemp = Uhat(1024x3840) @ Vhat^T + Wnorm_b + AmatBias  (f32)
// 128x128 block tile, 4 waves, 64x64 wave tile = 4x4 WMMA frags.
// Operands are L2-resident (15 MB << 192 MB), so direct global_load_b128
// feeds the WMMAs; no LDS staging and no prefetch needed.
// ---------------------------------------------------------------------------
__global__ void __launch_bounds__(128)
wmma_gemm_kernel(const _Float16* __restrict__ U, const _Float16* __restrict__ V,
                 const float* __restrict__ Abias, const float* __restrict__ Wnorm_b,
                 float* __restrict__ At)
{
    const int lane = threadIdx.x & 31;
    const int wave = threadIdx.x >> 5;
    const int lrow = lane & 15;
    const int lhi  = lane >> 4;          // 0/1: K-half selector
    const int row0 = blockIdx.y * 128 + (wave >> 1) * 64;
    const int col0 = blockIdx.x * 128 + (wave & 1) * 64;

    const _Float16* pa[4];
    const _Float16* pb[4];
    #pragma unroll
    for (int i = 0; i < 4; ++i) {
        pa[i] = U + (size_t)(row0 + i * 16 + lrow) * KDIM;   // A rows of Uhat
        pb[i] = V + (size_t)(col0 + i * 16 + lrow) * KDIM;   // B cols = rows of Vhat
    }

    const v8f vzero = {0.f, 0.f, 0.f, 0.f, 0.f, 0.f, 0.f, 0.f};
    v8f acc[4][4];
    #pragma unroll
    for (int i = 0; i < 4; ++i)
        #pragma unroll
        for (int j = 0; j < 4; ++j) acc[i][j] = vzero;

    for (int k0 = 0; k0 < KDIM; k0 += 32) {
        v16h af[4], bf[4];
        #pragma unroll
        for (int i = 0; i < 4; ++i) {
            // A 16x32 f16 layout: lane<16 -> K {0..7,16..23}; lane>=16 -> +8
            const v8h alo = *(const v8h*)(pa[i] + k0 + 8 * lhi);
            const v8h ahi = *(const v8h*)(pa[i] + k0 + 16 + 8 * lhi);
            af[i] = __builtin_shufflevector(alo, ahi,
                      0,1,2,3,4,5,6,7,8,9,10,11,12,13,14,15);
            // B 32x16 f16 layout: lane = column; lane<16 -> K {0..7,16..23}; lane>=16 -> +8
            const v8h blo = *(const v8h*)(pb[i] + k0 + 8 * lhi);
            const v8h bhi = *(const v8h*)(pb[i] + k0 + 16 + 8 * lhi);
            bf[i] = __builtin_shufflevector(blo, bhi,
                      0,1,2,3,4,5,6,7,8,9,10,11,12,13,14,15);
        }
        #pragma unroll
        for (int i = 0; i < 4; ++i)
            #pragma unroll
            for (int j = 0; j < 4; ++j)
                acc[i][j] = __builtin_amdgcn_wmma_f32_16x16x32_f16(
                    /*neg_a=*/false, af[i], /*neg_b=*/false, bf[j],
                    /*c_mod=*/(short)0, acc[i][j],
                    /*reuse_a=*/false, /*reuse_b=*/false);
    }

    const float bn = Wnorm_b[0];
    #pragma unroll
    for (int i = 0; i < 4; ++i)
        #pragma unroll
        for (int j = 0; j < 4; ++j)
            #pragma unroll
            for (int r = 0; r < 8; ++r) {
                const int row = row0 + i * 16 + r + 8 * lhi;
                const int col = col0 + j * 16 + lrow;
                const size_t idx = (size_t)row * NN + col;
                At[idx] = acc[i][j][r] + bn + Abias[idx];
            }
}

// ---------------------------------------------------------------------------
// Stage C: out[i,j] = (i==j) ? 0 : sigmoid(s*At[i,j] + (1-s)*At[j,i])
// ---------------------------------------------------------------------------
__global__ void __launch_bounds__(256)
mix_sigmoid_kernel(const float* __restrict__ At,
                   const float* __restrict__ scalar_a,
                   float* __restrict__ out)
{
    const int idx = blockIdx.x * 256 + threadIdx.x;
    if (idx >= NN * NN) return;
    const int i = idx >> 10;
    const int j = idx & (NN - 1);
    const float s = 1.0f / (1.0f + expf(-scalar_a[0]));
    const float a = At[idx];
    const float b = At[(size_t)j * NN + i];
    const float m = s * a + (1.0f - s) * b;
    out[idx] = (i == j) ? 0.0f : 1.0f / (1.0f + expf(-m));
}

// ---------------------------------------------------------------------------
extern "C" void kernel_launch(void* const* d_in, const int* in_sizes, int n_in,
                              void* d_out, int out_size, void* d_ws, size_t ws_size,
                              hipStream_t stream)
{
    const float* x        = (const float*)d_in[0];
    const float* Wu_w     = (const float*)d_in[1];
    const float* Wu_b     = (const float*)d_in[2];
    const float* Wv_w     = (const float*)d_in[3];
    const float* Wv_b     = (const float*)d_in[4];
    const float* Wnorm_w  = (const float*)d_in[5];
    const float* Wnorm_b  = (const float*)d_in[6];
    const float* scalar_a = (const float*)d_in[7];
    const float* Abias    = (const float*)d_in[8];
    float* out = (float*)d_out;

    // workspace layout: Uhat f16 | Vhat f16 | Atemp f32  (~19 MB)
    _Float16* Uh = (_Float16*)d_ws;
    _Float16* Vh = Uh + (size_t)NN * KDIM;
    float*    At = (float*)(Vh + (size_t)NN * KDIM);

    proj_norm_kernel<<<NN, 128, 0, stream>>>(x, Wu_w, Wu_b, Wv_w, Wv_b, Wnorm_w, Uh, Vh);

    dim3 grid(NN / 128, NN / 128);   // 8x8 blocks, 128x128 tile each
    wmma_gemm_kernel<<<grid, 128, 0, stream>>>(Uh, Vh, Abias, Wnorm_b, At);

    mix_sigmoid_kernel<<<(NN * NN + 255) / 256, 256, 0, stream>>>(At, scalar_a, out);
}
